// SimpleAttention_67379446940103
// MI455X (gfx1250) — matrix operand order repaired and checked by reading, not compiled, not run on hardware
//
#include <hip/hip_runtime.h>
#include <hip/hip_bf16.h>

typedef __bf16 bf16_t;
typedef __attribute__((ext_vector_type(16))) __bf16 v16bf;
typedef __attribute__((ext_vector_type(8)))  __bf16 v8bf;
typedef __attribute__((ext_vector_type(8)))  float  v8f;

#define WMMA_BF16(A, B, C) \
    __builtin_amdgcn_wmma_f32_16x16x32_bf16(false, (A), false, (B), (short)0, (C), false, false)

// ---------------------------------------------------------------------------
// Fragment loaders (wave32, CDNA5 VGPR layouts per ISA 7.12.2)
// A operand (16x32, 16-bit): lane<16 holds row (lane&15), K = {0..7, 16..23}
//                            lane>=16 holds row (lane&15), K = {8..15, 24..31}
// caller passes base = row_ptr + kbase + (lane>>4)*8
__device__ __forceinline__ v16bf ld_frag_a(const bf16_t* p) {
    const v8bf* q = (const v8bf*)p;
    v8bf lo = q[0];       // K +0..7
    v8bf hi = q[2];       // K +16..23
    return __builtin_shufflevector(lo, hi, 0,1,2,3,4,5,6,7,8,9,10,11,12,13,14,15);
}
// B operand (32x16, 16-bit): lane<16 holds col (lane&15), K = {0..7, 16..23};
//                            lane>=16 holds col (lane&15), K = {8..15, 24..31}.
// caller passes base = colrow_ptr + kbase + (lane>>4)*8  (16B aligned)
__device__ __forceinline__ v16bf ld_frag_b(const bf16_t* p) {
    const v8bf* q = (const v8bf*)p;
    v8bf lo = q[0];
    v8bf hi = q[2];
    return __builtin_shufflevector(lo, hi, 0,1,2,3,4,5,6,7,8,9,10,11,12,13,14,15);
}

// ---------------------------------------------------------------------------
// Conversion / transpose kernels
__global__ void cvt_f32_to_bf16(const float* __restrict__ in, bf16_t* __restrict__ out, int n) {
    int i = (blockIdx.x * blockDim.x + threadIdx.x) * 4;
    if (i + 3 < n) {
        float4 f = *(const float4*)(in + i);
        out[i + 0] = (bf16_t)f.x;
        out[i + 1] = (bf16_t)f.y;
        out[i + 2] = (bf16_t)f.z;
        out[i + 3] = (bf16_t)f.w;
    }
}

// in:  [K=768, N] row-major f32 ;  out: [N, 768] row-major bf16 (B-operand layout)
__global__ void transpose_cvt(const float* __restrict__ in, bf16_t* __restrict__ out, int ncols) {
    int o = blockIdx.x * blockDim.x + threadIdx.x;
    if (o >= 768 * ncols) return;
    int n = o / 768;
    int k = o - n * 768;
    out[o] = (bf16_t)in[(size_t)k * ncols + n];
}

// ---------------------------------------------------------------------------
// QKV GEMM: [8192,768]bf16 x [768,2304] -> scatter Q,K as [BH,2048,64], V as [BH,64,2048]
__global__ __launch_bounds__(128) void qkv_gemm(
    const bf16_t* __restrict__ A, const bf16_t* __restrict__ Bt,
    bf16_t* __restrict__ qb, bf16_t* __restrict__ kb, bf16_t* __restrict__ vb) {
    const int lane = threadIdx.x & 31;
    const int wave = threadIdx.x >> 5;
    const int l15 = lane & 15, hi = lane >> 4;
    const int mbase = blockIdx.x * 64 + (wave >> 1) * 32;
    const int nbase = blockIdx.y * 128 + (wave & 1) * 64;

    v8f acc[2][4] = {};
    for (int kk = 0; kk < 768; kk += 32) {
        v16bf a[2], b[4];
#pragma unroll
        for (int mi = 0; mi < 2; ++mi)
            a[mi] = ld_frag_a(A + (size_t)(mbase + 16 * mi + l15) * 768 + kk + hi * 8);
#pragma unroll
        for (int ni = 0; ni < 4; ++ni)
            b[ni] = ld_frag_b(Bt + (size_t)(nbase + 16 * ni + l15) * 768 + kk + hi * 8);
#pragma unroll
        for (int mi = 0; mi < 2; ++mi)
#pragma unroll
            for (int ni = 0; ni < 4; ++ni)
                acc[mi][ni] = WMMA_BF16(a[mi], b[ni], acc[mi][ni]);
    }
#pragma unroll
    for (int mi = 0; mi < 2; ++mi) {
#pragma unroll
        for (int ni = 0; ni < 4; ++ni) {
            const int col0  = nbase + 16 * ni;
            const int which = col0 / 768;          // 0=Q 1=K 2=V (128-col block never straddles)
            const int hcol  = col0 - which * 768;
            const int h     = hcol >> 6;
            const int hd    = (hcol & 63) + l15;
#pragma unroll
            for (int r = 0; r < 8; ++r) {
                const int m  = mbase + 16 * mi + r + 8 * hi;
                const int b_ = m >> 11, s = m & 2047;
                const int bh = b_ * 12 + h;
                bf16_t val = (bf16_t)acc[mi][ni][r];
                if (which == 0)      qb[((size_t)bh * 2048 + s) * 64 + hd] = val;
                else if (which == 1) kb[((size_t)bh * 2048 + s) * 64 + hd] = val;
                else                 vb[((size_t)bh * 64 + hd) * 2048 + s] = val;  // V^T
            }
        }
    }
}

// ---------------------------------------------------------------------------
// Flash attention. One wave = 16 queries. S^T = K*Q^T so softmax is lane-local;
// S^T C-fragments re-pack directly as A-operand of P*V (no LDS, no transpose).
__global__ __launch_bounds__(128) void flash_attn(
    const bf16_t* __restrict__ qbuf, const bf16_t* __restrict__ kbuf,
    const bf16_t* __restrict__ vbuf, bf16_t* __restrict__ obuf) {
    const int lane = threadIdx.x & 31;
    const int wave = threadIdx.x >> 5;
    const int l15 = lane & 15, hi = lane >> 4;
    const int bh = blockIdx.y;
    const int qbase = blockIdx.x * 64 + wave * 16;

    const bf16_t* q = qbuf + (size_t)bh * 2048 * 64;
    const bf16_t* k = kbuf + (size_t)bh * 2048 * 64;
    const bf16_t* v = vbuf + (size_t)bh * 64 * 2048;   // [64, 2048] = V^T

    v16bf qf[2];
#pragma unroll
    for (int kk = 0; kk < 2; ++kk)
        qf[kk] = ld_frag_b(q + (size_t)(qbase + l15) * 64 + kk * 32 + hi * 8);

    const int myq = qbase + l15;          // query owned by this lane (both halves)
    float m = -INFINITY, lsum = 0.f;
    v8f o[4] = {};

    const int nfull = qbase >> 6;          // diag tile index; tiles < nfull are unmasked
    for (int kt = 0; kt <= nfull; ++kt) {
        const int kbase = kt * 64;
        const bool masked = (kt == nfull);

        // S^T tile [64 keys x 16 queries]: 4 m-steps x 2 k-steps
        v8f st[4];
#pragma unroll
        for (int f = 0; f < 4; ++f) {
            v8f c = {};
#pragma unroll
            for (int kk = 0; kk < 2; ++kk) {
                v16bf a = ld_frag_a(k + (size_t)(kbase + 16 * f + l15) * 64 + kk * 32 + hi * 8);
                c = WMMA_BF16(a, qf[kk], c);
            }
            st[f] = c;
        }
        // scale + causal mask + tile max
        float tmax = -INFINITY;
#pragma unroll
        for (int f = 0; f < 4; ++f)
#pragma unroll
            for (int r = 0; r < 8; ++r) {
                float s = st[f][r] * 0.125f;                  // 1/sqrt(64)
                if (masked) {
                    int key = kbase + 16 * f + 8 * hi + r;
                    if (key > myq) s = -INFINITY;
                }
                st[f][r] = s;
                tmax = fmaxf(tmax, s);
            }
        tmax = fmaxf(tmax, __shfl_xor(tmax, 16, 32));
        const float mnew = fmaxf(m, tmax);
        const float corr = __expf(m - mnew);                  // 0 on first tile
        float ps = 0.f;
#pragma unroll
        for (int f = 0; f < 4; ++f)
#pragma unroll
            for (int r = 0; r < 8; ++r) {
                float p = __expf(st[f][r] - mnew);
                st[f][r] = p;
                ps += p;
            }
        ps += __shfl_xor(ps, 16, 32);
        lsum = lsum * corr + ps;
        m = mnew;

        // rescale O accumulators: O element r belongs to query 8*hi + r
#pragma unroll
        for (int r = 0; r < 8; ++r) {
            float cr = __shfl(corr, 8 * hi + r, 32);
#pragma unroll
            for (int n = 0; n < 4; ++n) o[n][r] *= cr;
        }

        // re-pack P (S^T frags) as A operand of P[16 x 64keys] @ V[64keys x 64hd]
        v16bf pa[2];
#pragma unroll
        for (int t = 0; t < 2; ++t) {
#pragma unroll
            for (int e = 0; e < 8; ++e) {
                pa[t][e]     = (bf16_t)st[2 * t][e];
                pa[t][e + 8] = (bf16_t)st[2 * t + 1][e];
            }
        }
#pragma unroll
        for (int n = 0; n < 4; ++n)
#pragma unroll
            for (int t = 0; t < 2; ++t) {
                v16bf bv = ld_frag_b(v + (size_t)(16 * n + l15) * 2048 + kbase + 32 * t + hi * 8);
                o[n] = WMMA_BF16(pa[t], bv, o[n]);
            }
    }

    // epilogue: normalize, write [token, H] bf16 for the output projection
    const int h = bh % 12, b_ = bh / 12;
#pragma unroll
    for (int r = 0; r < 8; ++r) {
        float lr  = __shfl(lsum, 8 * hi + r, 32);
        float inv = 1.0f / lr;
        int row = b_ * 2048 + qbase + 8 * hi + r;
#pragma unroll
        for (int n = 0; n < 4; ++n)
            obuf[(size_t)row * 768 + h * 64 + 16 * n + l15] = (bf16_t)(o[n][r] * inv);
    }
}

// ---------------------------------------------------------------------------
// Output projection: [8192,768]bf16 x [768,768] -> d_out f32
__global__ __launch_bounds__(128) void proj_gemm(
    const bf16_t* __restrict__ A, const bf16_t* __restrict__ Bt, float* __restrict__ out) {
    const int lane = threadIdx.x & 31;
    const int wave = threadIdx.x >> 5;
    const int l15 = lane & 15, hi = lane >> 4;
    const int mbase = blockIdx.x * 64 + (wave >> 1) * 32;
    const int nbase = blockIdx.y * 128 + (wave & 1) * 64;

    v8f acc[2][4] = {};
    for (int kk = 0; kk < 768; kk += 32) {
        v16bf a[2], b[4];
#pragma unroll
        for (int mi = 0; mi < 2; ++mi)
            a[mi] = ld_frag_a(A + (size_t)(mbase + 16 * mi + l15) * 768 + kk + hi * 8);
#pragma unroll
        for (int ni = 0; ni < 4; ++ni)
            b[ni] = ld_frag_b(Bt + (size_t)(nbase + 16 * ni + l15) * 768 + kk + hi * 8);
#pragma unroll
        for (int mi = 0; mi < 2; ++mi)
#pragma unroll
            for (int ni = 0; ni < 4; ++ni)
                acc[mi][ni] = WMMA_BF16(a[mi], b[ni], acc[mi][ni]);
    }
#pragma unroll
    for (int mi = 0; mi < 2; ++mi)
#pragma unroll
        for (int ni = 0; ni < 4; ++ni)
#pragma unroll
            for (int r = 0; r < 8; ++r) {
                int mrow = mbase + 16 * mi + r + 8 * hi;
                out[(size_t)mrow * 768 + nbase + 16 * ni + l15] = acc[mi][ni][r];
            }
}

// ---------------------------------------------------------------------------
extern "C" void kernel_launch(void* const* d_in, const int* in_sizes, int n_in,
                              void* d_out, int out_size, void* d_ws, size_t ws_size,
                              hipStream_t stream) {
    const float* x     = (const float*)d_in[0];   // [4,2048,768]
    const float* w_qkv = (const float*)d_in[1];   // [768,2304]
    const float* w_out = (const float*)d_in[2];   // [768,768]
    float* out = (float*)d_out;

    char* ws = (char*)d_ws;
    const size_t SZ_TOK = (size_t)8192 * 768 * sizeof(bf16_t);   // 12.58 MB
    bf16_t* xb     = (bf16_t*)(ws);                              // [8192,768]
    bf16_t* wqkvT  = (bf16_t*)(ws + SZ_TOK);                     // [2304,768]
    bf16_t* woutT  = (bf16_t*)(ws + SZ_TOK + 3538944);           // [768,768]
    bf16_t* qb     = (bf16_t*)(ws + SZ_TOK + 3538944 + 1179648);
    bf16_t* kb     = (bf16_t*)((char*)qb + SZ_TOK);              // [48,2048,64]
    bf16_t* vb     = (bf16_t*)((char*)kb + SZ_TOK);              // [48,64,2048] (V^T)
    bf16_t* ab     = (bf16_t*)((char*)vb + SZ_TOK);              // attn out [8192,768]

    cvt_f32_to_bf16<<<6144, 256, 0, stream>>>(x, xb, 8192 * 768);
    transpose_cvt<<<(768 * 2304 + 255) / 256, 256, 0, stream>>>(w_qkv, wqkvT, 2304);
    transpose_cvt<<<(768 * 768 + 255) / 256, 256, 0, stream>>>(w_out, woutT, 768);

    qkv_gemm<<<dim3(128, 18), 128, 0, stream>>>(xb, wqkvT, qb, kb, vb);
    flash_attn<<<dim3(32, 48), 128, 0, stream>>>(qb, kb, vb, ab);
    proj_gemm<<<dim3(128, 6), 128, 0, stream>>>(ab, woutT, out);
}
